// EquivariantBlock_52192442581789
// MI455X (gfx1250) — matrix operand order repaired and checked by reading, not compiled, not run on hardware
//
#include <hip/hip_runtime.h>
#include <hip/hip_bf16.h>
#include <math.h>

// ---------------------------------------------------------------------------
// EGNN EquivariantBlock for gfx1250 (MI455X).
// All matmuls -> v_wmma_f32_16x16x32_f16 (f16 in, f32 accum).
// ---------------------------------------------------------------------------

typedef _Float16 vh16 __attribute__((ext_vector_type(16)));
typedef _Float16 vh8  __attribute__((ext_vector_type(8)));
typedef float    vf8  __attribute__((ext_vector_type(8)));

#define HDIM 128
#define TE   64      // rows (edges or nodes) per workgroup tile
#define ASTRIDE 272  // LDS halves per row of the 256-wide A tile (padded)
#define MSTRIDE 136  // LDS halves per row of the 128-wide mid tile (padded)

__device__ __forceinline__ float silu_f(float v) {
    return v / (1.0f + __expf(-v));
}

// A fragment (16x32 f16) for one 16-row tile from an LDS row-major buffer.
// Lane layout (wave32): row M = lane%16, g = lane/16.
//   elems j=0..7  -> K = kk + 8g + j        (16 contiguous bytes)
//   elems j=8..15 -> K = kk + 16 + 8g + (j-8)
__device__ __forceinline__ vh16 load_a_frag(const _Float16* base, int stride,
                                            int row, int kk, int g) {
    const _Float16* p = base + row * stride + kk + 8 * g;
    vh8 lo = *(const vh8*)p;
    vh8 hi = *(const vh8*)(p + 16);
    vh16 a;
#pragma unroll
    for (int i = 0; i < 8; i++) { a[i] = lo[i]; a[8 + i] = hi[i]; }
    return a;
}

// B fragment (32x16 f16) from a transposed weight matrix wT[N][Ks] (f16).
// Lane holds column N, elems j=0..7 -> K = kk + 8g + j, j=8..15 -> K = kk + 16 + 8g + (j-8).
__device__ __forceinline__ vh16 load_b_frag(const _Float16* wT, int Ks,
                                            int n, int kk, int g) {
    const _Float16* p = wT + (size_t)n * Ks + kk + 8 * g;
    vh8 lo = *(const vh8*)p;
    vh8 hi = *(const vh8*)(p + 16);
    vh16 b;
#pragma unroll
    for (int i = 0; i < 8; i++) { b[i] = lo[i]; b[8 + i] = hi[i]; }
    return b;
}

// ---------------------------------------------------------------------------
// Prep kernels
// ---------------------------------------------------------------------------
__global__ void prep_edges_kernel(const float* __restrict__ x,
                                  const int* __restrict__ ei,
                                  float* __restrict__ radial,
                                  float* __restrict__ cdiff, int E) {
    int e = blockIdx.x * blockDim.x + threadIdx.x;
    if (e >= E) return;
    int r = ei[e], c = ei[E + e];
    float dx = x[r * 3 + 0] - x[c * 3 + 0];
    float dy = x[r * 3 + 1] - x[c * 3 + 1];
    float dz = x[r * 3 + 2] - x[c * 3 + 2];
    float rad = dx * dx + dy * dy + dz * dz;
    float inv = 1.0f / (sqrtf(rad + 1e-8f) + 1.0f);   // NORM_CONST = 1
    radial[e] = rad;
    cdiff[e * 3 + 0] = dx * inv;
    cdiff[e * 3 + 1] = dy * inv;
    cdiff[e * 3 + 2] = dz * inv;
}

// src is [K][128] f32 (row-major), dst is [128][K] f16 (transposed)
__global__ void transpose_f16_kernel(const float* __restrict__ src,
                                     _Float16* __restrict__ dst, int K) {
    int i = blockIdx.x * blockDim.x + threadIdx.x;
    if (i >= 128 * K) return;
    int n = i / K, k = i % K;
    dst[(size_t)n * K + k] = (_Float16)src[(size_t)k * 128 + n];
}

__global__ void finalize_x_kernel(const float* __restrict__ x,
                                  const float* __restrict__ xacc,
                                  float* __restrict__ xout, int n) {
    int i = blockIdx.x * blockDim.x + threadIdx.x;
    if (i < n) xout[i] = x[i] + xacc[i] * 0.01f;   // /NORM
}

// ---------------------------------------------------------------------------
// GCL edge kernel: mij = silu(silu([h_r,h_c,ea]@W1+b1)@W2+b2); att=sigmoid(..);
// scatter mij*att into agg[row] (raw sum; /100 applied in node kernel).
// ---------------------------------------------------------------------------
__global__ __launch_bounds__(256) void egcl_edge_kernel(
    const float* __restrict__ h, const int* __restrict__ ei,
    const float* __restrict__ radial, const float* __restrict__ eattr,
    const _Float16* __restrict__ ew1t, const float* __restrict__ ew1tail,
    const float* __restrict__ eb1,
    const _Float16* __restrict__ ew2t, const float* __restrict__ eb2,
    const float* __restrict__ aw, const float* __restrict__ ab,
    float* __restrict__ agg, int E) {
    __shared__ __align__(16) _Float16 Ah[TE * ASTRIDE];
    __shared__ __align__(16) _Float16 Mb[TE * MSTRIDE];
    __shared__ float sEA0[TE], sEA1[TE], sAtt[TE];
    __shared__ int sRow[TE];

    const int t = threadIdx.x;
    const int te = blockIdx.x * TE;

    // ---- phase 0: gather h[row]||h[col] -> f16 LDS A tile ----
    {
        int e = t >> 2, p = t & 3;
        int ge = te + e;
        _Float16* dr = Ah + e * ASTRIDE + p * 32;
        if (ge < E) {
            int r = ei[ge], c = ei[E + ge];
            const float4* hr = (const float4*)(h + (size_t)r * HDIM) + p * 8;
            const float4* hc = (const float4*)(h + (size_t)c * HDIM) + p * 8;
#pragma unroll
            for (int i = 0; i < 8; i++) {
                float4 a = hr[i], b = hc[i];
                dr[i * 4 + 0] = (_Float16)a.x; dr[i * 4 + 1] = (_Float16)a.y;
                dr[i * 4 + 2] = (_Float16)a.z; dr[i * 4 + 3] = (_Float16)a.w;
                dr[128 + i * 4 + 0] = (_Float16)b.x; dr[128 + i * 4 + 1] = (_Float16)b.y;
                dr[128 + i * 4 + 2] = (_Float16)b.z; dr[128 + i * 4 + 3] = (_Float16)b.w;
            }
            if (p == 0) { sEA0[e] = radial[ge]; sEA1[e] = eattr[ge]; sRow[e] = r; }
        } else {
#pragma unroll
            for (int i = 0; i < 32; i++) { dr[i] = (_Float16)0.0f; dr[128 + i] = (_Float16)0.0f; }
            if (p == 0) { sEA0[e] = 0.0f; sEA1[e] = 0.0f; sRow[e] = -1; }
        }
    }
    __syncthreads();

    const int ln = t & 15;
    const int g  = (t >> 4) & 1;
    const int n  = ((t >> 5) << 4) + ln;   // output column owned by this lane

    // ---- phase 1: GEMM1 (K=256) + bias + ea rank-2 tail + SiLU -> Mb ----
    vf8 acc[4];
#pragma unroll
    for (int rt = 0; rt < 4; rt++)
#pragma unroll
        for (int i = 0; i < 8; i++) acc[rt][i] = 0.0f;

#pragma unroll
    for (int kk = 0; kk < 256; kk += 32) {
        vh16 b = load_b_frag(ew1t, 256, n, kk, g);
#pragma unroll
        for (int rt = 0; rt < 4; rt++) {
            vh16 a = load_a_frag(Ah, ASTRIDE, rt * 16 + ln, kk, g);
            acc[rt] = __builtin_amdgcn_wmma_f32_16x16x32_f16(
                false, a, false, b, (short)0, acc[rt], false, false);
        }
    }
    {
        float b1 = eb1[n], w6 = ew1tail[n], w7 = ew1tail[HDIM + n];
#pragma unroll
        for (int rt = 0; rt < 4; rt++) {
#pragma unroll
            for (int i = 0; i < 8; i++) {
                int m = rt * 16 + i + 8 * g;
                float v = acc[rt][i] + b1 + sEA0[m] * w6 + sEA1[m] * w7;
                Mb[m * MSTRIDE + n] = (_Float16)silu_f(v);
            }
        }
    }
    __syncthreads();

    // ---- phase 2: GEMM2 (K=128) + bias + SiLU; keep f32, stash f16 ----
    vf8 acc2[4];
#pragma unroll
    for (int rt = 0; rt < 4; rt++)
#pragma unroll
        for (int i = 0; i < 8; i++) acc2[rt][i] = 0.0f;

#pragma unroll
    for (int kk = 0; kk < 128; kk += 32) {
        vh16 b = load_b_frag(ew2t, 128, n, kk, g);
#pragma unroll
        for (int rt = 0; rt < 4; rt++) {
            vh16 a = load_a_frag(Mb, MSTRIDE, rt * 16 + ln, kk, g);
            acc2[rt] = __builtin_amdgcn_wmma_f32_16x16x32_f16(
                false, a, false, b, (short)0, acc2[rt], false, false);
        }
    }
    {
        float b2 = eb2[n];
#pragma unroll
        for (int rt = 0; rt < 4; rt++) {
#pragma unroll
            for (int i = 0; i < 8; i++) {
                int m = rt * 16 + i + 8 * g;
                float v = silu_f(acc2[rt][i] + b2);
                acc2[rt][i] = v;
                Ah[m * MSTRIDE + n] = (_Float16)v;   // reuse Ah as m2 buffer
            }
        }
    }
    __syncthreads();

    // ---- phase 3: attention att[e] = sigmoid(dot(m2[e,:], aw) + ab) ----
    {
        int e = t >> 2, p = t & 3;
        const _Float16* mp = Ah + e * MSTRIDE + p * 32;
        const float* ap = aw + p * 32;
        float s = 0.0f;
#pragma unroll
        for (int i = 0; i < 32; i++) s += (float)mp[i] * ap[i];
        s += __shfl_xor(s, 1, 4);
        s += __shfl_xor(s, 2, 4);
        if (p == 0) sAtt[e] = 1.0f / (1.0f + __expf(-(s + ab[0])));
    }
    __syncthreads();

    // ---- phase 4: scatter m2*att into agg[row] ----
#pragma unroll
    for (int rt = 0; rt < 4; rt++) {
#pragma unroll
        for (int i = 0; i < 8; i++) {
            int m = rt * 16 + i + 8 * g;
            int r = sRow[m];
            if (r >= 0)
                atomicAdd(agg + (size_t)r * HDIM + n, acc2[rt][i] * sAtt[m]);
        }
    }
}

// ---------------------------------------------------------------------------
// GCL node kernel: h_out = h + silu([h, agg/100]@NW1+nb1)@NW2 + nb2
// ---------------------------------------------------------------------------
__global__ __launch_bounds__(256) void egcl_node_kernel(
    const float* __restrict__ h_in, const float* __restrict__ agg,
    const _Float16* __restrict__ nw1t, const float* __restrict__ nb1,
    const _Float16* __restrict__ nw2t, const float* __restrict__ nb2,
    float* __restrict__ h_out, int V) {
    __shared__ __align__(16) _Float16 Ah[TE * ASTRIDE];
    __shared__ __align__(16) _Float16 Mb[TE * MSTRIDE];

    const int t = threadIdx.x;
    const int vb = blockIdx.x * TE;

    {
        int e = t >> 2, p = t & 3;
        int v = vb + e;
        _Float16* dr = Ah + e * ASTRIDE + p * 32;
        if (v < V) {
            const float4* hv = (const float4*)(h_in + (size_t)v * HDIM) + p * 8;
            const float4* av = (const float4*)(agg + (size_t)v * HDIM) + p * 8;
#pragma unroll
            for (int i = 0; i < 8; i++) {
                float4 a = hv[i], b = av[i];
                dr[i * 4 + 0] = (_Float16)a.x; dr[i * 4 + 1] = (_Float16)a.y;
                dr[i * 4 + 2] = (_Float16)a.z; dr[i * 4 + 3] = (_Float16)a.w;
                dr[128 + i * 4 + 0] = (_Float16)(b.x * 0.01f);
                dr[128 + i * 4 + 1] = (_Float16)(b.y * 0.01f);
                dr[128 + i * 4 + 2] = (_Float16)(b.z * 0.01f);
                dr[128 + i * 4 + 3] = (_Float16)(b.w * 0.01f);
            }
        } else {
#pragma unroll
            for (int i = 0; i < 32; i++) { dr[i] = (_Float16)0.0f; dr[128 + i] = (_Float16)0.0f; }
        }
    }
    __syncthreads();

    const int ln = t & 15;
    const int g  = (t >> 4) & 1;
    const int n  = ((t >> 5) << 4) + ln;

    vf8 acc[4];
#pragma unroll
    for (int rt = 0; rt < 4; rt++)
#pragma unroll
        for (int i = 0; i < 8; i++) acc[rt][i] = 0.0f;

#pragma unroll
    for (int kk = 0; kk < 256; kk += 32) {
        vh16 b = load_b_frag(nw1t, 256, n, kk, g);
#pragma unroll
        for (int rt = 0; rt < 4; rt++) {
            vh16 a = load_a_frag(Ah, ASTRIDE, rt * 16 + ln, kk, g);
            acc[rt] = __builtin_amdgcn_wmma_f32_16x16x32_f16(
                false, a, false, b, (short)0, acc[rt], false, false);
        }
    }
    {
        float b1 = nb1[n];
#pragma unroll
        for (int rt = 0; rt < 4; rt++)
#pragma unroll
            for (int i = 0; i < 8; i++) {
                int m = rt * 16 + i + 8 * g;
                Mb[m * MSTRIDE + n] = (_Float16)silu_f(acc[rt][i] + b1);
            }
    }
    __syncthreads();

    vf8 acc2[4];
#pragma unroll
    for (int rt = 0; rt < 4; rt++)
#pragma unroll
        for (int i = 0; i < 8; i++) acc2[rt][i] = 0.0f;

#pragma unroll
    for (int kk = 0; kk < 128; kk += 32) {
        vh16 b = load_b_frag(nw2t, 128, n, kk, g);
#pragma unroll
        for (int rt = 0; rt < 4; rt++) {
            vh16 a = load_a_frag(Mb, MSTRIDE, rt * 16 + ln, kk, g);
            acc2[rt] = __builtin_amdgcn_wmma_f32_16x16x32_f16(
                false, a, false, b, (short)0, acc2[rt], false, false);
        }
    }
    {
        float b2 = nb2[n];
#pragma unroll
        for (int rt = 0; rt < 4; rt++)
#pragma unroll
            for (int i = 0; i < 8; i++) {
                int m = rt * 16 + i + 8 * g;
                int v = vb + m;
                if (v < V) {
                    size_t idx = (size_t)v * HDIM + n;
                    h_out[idx] = h_in[idx] + acc2[rt][i] + b2;   // residual, no act
                }
            }
    }
}

// ---------------------------------------------------------------------------
// Equivariant coord kernel: s = silu(silu(inp@CW1+cb1)@CW2+cb2)@cw3;
// scatter cdiff*s into xacc[row].
// ---------------------------------------------------------------------------
__global__ __launch_bounds__(256) void coord_kernel(
    const float* __restrict__ h, const int* __restrict__ ei,
    const float* __restrict__ radial, const float* __restrict__ eattr,
    const float* __restrict__ cdiff,
    const _Float16* __restrict__ cw1t, const float* __restrict__ cw1tail,
    const float* __restrict__ cb1,
    const _Float16* __restrict__ cw2t, const float* __restrict__ cb2,
    const float* __restrict__ cw3,
    float* __restrict__ xacc, int E) {
    __shared__ __align__(16) _Float16 Ah[TE * ASTRIDE];
    __shared__ __align__(16) _Float16 Mb[TE * MSTRIDE];
    __shared__ float sEA0[TE], sEA1[TE], sS[TE];
    __shared__ int sRow[TE];

    const int t = threadIdx.x;
    const int te = blockIdx.x * TE;

    {
        int e = t >> 2, p = t & 3;
        int ge = te + e;
        _Float16* dr = Ah + e * ASTRIDE + p * 32;
        if (ge < E) {
            int r = ei[ge], c = ei[E + ge];
            const float4* hr = (const float4*)(h + (size_t)r * HDIM) + p * 8;
            const float4* hc = (const float4*)(h + (size_t)c * HDIM) + p * 8;
#pragma unroll
            for (int i = 0; i < 8; i++) {
                float4 a = hr[i], b = hc[i];
                dr[i * 4 + 0] = (_Float16)a.x; dr[i * 4 + 1] = (_Float16)a.y;
                dr[i * 4 + 2] = (_Float16)a.z; dr[i * 4 + 3] = (_Float16)a.w;
                dr[128 + i * 4 + 0] = (_Float16)b.x; dr[128 + i * 4 + 1] = (_Float16)b.y;
                dr[128 + i * 4 + 2] = (_Float16)b.z; dr[128 + i * 4 + 3] = (_Float16)b.w;
            }
            if (p == 0) { sEA0[e] = radial[ge]; sEA1[e] = eattr[ge]; sRow[e] = r; }
        } else {
#pragma unroll
            for (int i = 0; i < 32; i++) { dr[i] = (_Float16)0.0f; dr[128 + i] = (_Float16)0.0f; }
            if (p == 0) { sEA0[e] = 0.0f; sEA1[e] = 0.0f; sRow[e] = -1; }
        }
    }
    __syncthreads();

    const int ln = t & 15;
    const int g  = (t >> 4) & 1;
    const int n  = ((t >> 5) << 4) + ln;

    vf8 acc[4];
#pragma unroll
    for (int rt = 0; rt < 4; rt++)
#pragma unroll
        for (int i = 0; i < 8; i++) acc[rt][i] = 0.0f;

#pragma unroll
    for (int kk = 0; kk < 256; kk += 32) {
        vh16 b = load_b_frag(cw1t, 256, n, kk, g);
#pragma unroll
        for (int rt = 0; rt < 4; rt++) {
            vh16 a = load_a_frag(Ah, ASTRIDE, rt * 16 + ln, kk, g);
            acc[rt] = __builtin_amdgcn_wmma_f32_16x16x32_f16(
                false, a, false, b, (short)0, acc[rt], false, false);
        }
    }
    {
        float b1 = cb1[n], w6 = cw1tail[n], w7 = cw1tail[HDIM + n];
#pragma unroll
        for (int rt = 0; rt < 4; rt++)
#pragma unroll
            for (int i = 0; i < 8; i++) {
                int m = rt * 16 + i + 8 * g;
                float v = acc[rt][i] + b1 + sEA0[m] * w6 + sEA1[m] * w7;
                Mb[m * MSTRIDE + n] = (_Float16)silu_f(v);
            }
    }
    __syncthreads();

    vf8 acc2[4];
#pragma unroll
    for (int rt = 0; rt < 4; rt++)
#pragma unroll
        for (int i = 0; i < 8; i++) acc2[rt][i] = 0.0f;

#pragma unroll
    for (int kk = 0; kk < 128; kk += 32) {
        vh16 b = load_b_frag(cw2t, 128, n, kk, g);
#pragma unroll
        for (int rt = 0; rt < 4; rt++) {
            vh16 a = load_a_frag(Mb, MSTRIDE, rt * 16 + ln, kk, g);
            acc2[rt] = __builtin_amdgcn_wmma_f32_16x16x32_f16(
                false, a, false, b, (short)0, acc2[rt], false, false);
        }
    }
    {
        float b2 = cb2[n];
#pragma unroll
        for (int rt = 0; rt < 4; rt++)
#pragma unroll
            for (int i = 0; i < 8; i++) {
                int m = rt * 16 + i + 8 * g;
                Ah[m * MSTRIDE + n] = (_Float16)silu_f(acc2[rt][i] + b2);
            }
    }
    __syncthreads();

    // s[e] = dot(t2[e,:], cw3)  (no bias, no activation)
    {
        int e = t >> 2, p = t & 3;
        const _Float16* mp = Ah + e * MSTRIDE + p * 32;
        const float* cp = cw3 + p * 32;
        float s = 0.0f;
#pragma unroll
        for (int i = 0; i < 32; i++) s += (float)mp[i] * cp[i];
        s += __shfl_xor(s, 1, 4);
        s += __shfl_xor(s, 2, 4);
        if (p == 0) sS[e] = s;
    }
    __syncthreads();

    {
        int e = t >> 2, d = t & 3;
        int ge = te + e;
        int r = sRow[e];
        if (d < 3 && r >= 0)
            atomicAdd(xacc + (size_t)r * 3 + d, cdiff[(size_t)ge * 3 + d] * sS[e]);
    }
}

// ---------------------------------------------------------------------------
// Host-side launch
// ---------------------------------------------------------------------------
extern "C" void kernel_launch(void* const* d_in, const int* in_sizes, int n_in,
                              void* d_out, int out_size, void* d_ws, size_t ws_size,
                              hipStream_t stream) {
    const float* h0    = (const float*)d_in[0];
    const float* x     = (const float*)d_in[1];
    const int*   ei    = (const int*)d_in[2];
    const float* eattr = (const float*)d_in[3];
    const float* ew1 = (const float*)d_in[4];
    const float* eb1 = (const float*)d_in[5];
    const float* ew2 = (const float*)d_in[6];
    const float* eb2 = (const float*)d_in[7];
    const float* aw  = (const float*)d_in[8];
    const float* ab  = (const float*)d_in[9];
    const float* nw1 = (const float*)d_in[10];
    const float* nb1 = (const float*)d_in[11];
    const float* nw2 = (const float*)d_in[12];
    const float* nb2 = (const float*)d_in[13];
    const float* cw1 = (const float*)d_in[14];
    const float* cb1 = (const float*)d_in[15];
    const float* cw2 = (const float*)d_in[16];
    const float* cb2 = (const float*)d_in[17];
    const float* cw3 = (const float*)d_in[18];

    const int V = in_sizes[0] / HDIM;
    const int E = in_sizes[3];
    const int NL = 2;
    (void)n_in; (void)out_size; (void)ws_size;

    // workspace carve-up (256B aligned)
    char* ws = (char*)d_ws;
    size_t off = 0;
    auto carve = [&](size_t bytes) -> char* {
        char* p = ws + off;
        off = (off + bytes + 255) & ~(size_t)255;
        return p;
    };
    float*    radial = (float*)carve((size_t)E * 4);
    float*    cdiff  = (float*)carve((size_t)E * 12);
    float*    agg    = (float*)carve((size_t)V * HDIM * 4);
    float*    xacc   = (float*)carve((size_t)V * 12);
    float*    hbuf   = (float*)carve((size_t)V * HDIM * 4);
    _Float16* ew1t   = (_Float16*)carve((size_t)NL * 128 * 256 * 2);
    _Float16* ew2t   = (_Float16*)carve((size_t)NL * 128 * 128 * 2);
    _Float16* nw1t   = (_Float16*)carve((size_t)NL * 128 * 256 * 2);
    _Float16* nw2t   = (_Float16*)carve((size_t)NL * 128 * 128 * 2);
    _Float16* cw1t   = (_Float16*)carve((size_t)128 * 256 * 2);
    _Float16* cw2t   = (_Float16*)carve((size_t)128 * 128 * 2);

    // ---- prep ----
    prep_edges_kernel<<<(E + 255) / 256, 256, 0, stream>>>(x, ei, radial, cdiff, E);
    const int gT256 = (128 * 256 + 255) / 256;
    const int gT128 = (128 * 128 + 255) / 256;
    for (int l = 0; l < NL; l++) {
        transpose_f16_kernel<<<gT256, 256, 0, stream>>>(ew1 + (size_t)l * 258 * 128,
                                                        ew1t + (size_t)l * 128 * 256, 256);
        transpose_f16_kernel<<<gT128, 256, 0, stream>>>(ew2 + (size_t)l * 128 * 128,
                                                        ew2t + (size_t)l * 128 * 128, 128);
        transpose_f16_kernel<<<gT256, 256, 0, stream>>>(nw1 + (size_t)l * 256 * 128,
                                                        nw1t + (size_t)l * 128 * 256, 256);
        transpose_f16_kernel<<<gT128, 256, 0, stream>>>(nw2 + (size_t)l * 128 * 128,
                                                        nw2t + (size_t)l * 128 * 128, 128);
    }
    transpose_f16_kernel<<<gT256, 256, 0, stream>>>(cw1, cw1t, 256);
    transpose_f16_kernel<<<gT128, 256, 0, stream>>>(cw2, cw2t, 128);

    float* h_final = (float*)d_out;                       // [V,128]
    float* x_out   = (float*)d_out + (size_t)V * HDIM;    // [V,3]
    float* houts[2] = { hbuf, h_final };

    const int nEdgeTiles = (E + TE - 1) / TE;
    const int nNodeTiles = (V + TE - 1) / TE;

    const float* hcur = h0;
    for (int l = 0; l < NL; l++) {
        hipMemsetAsync(agg, 0, (size_t)V * HDIM * 4, stream);
        egcl_edge_kernel<<<nEdgeTiles, 256, 0, stream>>>(
            hcur, ei, radial, eattr,
            ew1t + (size_t)l * 128 * 256,
            ew1 + (size_t)l * 258 * 128 + 256 * 128,   // tail rows 256,257 (f32)
            eb1 + (size_t)l * 128,
            ew2t + (size_t)l * 128 * 128, eb2 + (size_t)l * 128,
            aw + (size_t)l * 128, ab + l, agg, E);
        egcl_node_kernel<<<nNodeTiles, 256, 0, stream>>>(
            hcur, agg,
            nw1t + (size_t)l * 128 * 256, nb1 + (size_t)l * 128,
            nw2t + (size_t)l * 128 * 128, nb2 + (size_t)l * 128,
            houts[l], V);
        hcur = houts[l];
    }

    hipMemsetAsync(xacc, 0, (size_t)V * 12, stream);
    coord_kernel<<<nEdgeTiles, 256, 0, stream>>>(
        h_final, ei, radial, eattr, cdiff,
        cw1t, cw1 + 256 * 128, cb1, cw2t, cb2, cw3, xacc, E);
    finalize_x_kernel<<<(V * 3 + 255) / 256, 256, 0, stream>>>(x, xacc, x_out, V * 3);
}
